// update_e_13469017440644
// MI455X (gfx1250) — matrix operand order repaired and checked by reading, not compiled, not run on hardware
//
#include <hip/hip_runtime.h>

#define E_ 131072
#define T_ 524288
#define H_ 128
#define I_ 64

typedef __attribute__((ext_vector_type(16))) __bf16 v16bf;
typedef __attribute__((ext_vector_type(8)))  __bf16 v8bf;
typedef __attribute__((ext_vector_type(8)))  float  v8f;

static __device__ __forceinline__ __bf16 f2bf(float f) { return (__bf16)f; }
static __device__ __forceinline__ float silu_f(float v) {
    return v * __builtin_amdgcn_rcpf(1.f + __expf(-v));
}

// ---------------------------------------------------------------------------
// Transpose + bf16-convert a weight: Wt[n*KP + k] = bf16(W[k*N + n]), zero-pad K->KP
// ---------------------------------------------------------------------------
__global__ void conv_wt(const float* __restrict__ W, __bf16* __restrict__ Wt,
                        int K, int N, int KP)
{
    int i = blockIdx.x * blockDim.x + threadIdx.x;
    if (i >= N * KP) return;
    int n = i / KP, k = i - n * KP;
    Wt[i] = (k < K) ? f2bf(W[(size_t)k * N + n]) : f2bf(0.f);
}

// ---------------------------------------------------------------------------
// bf16-WMMA GEMM: out[M,N] = epi(A[M,K] (*premul) @ W[K,N] + bias)
// Weights pre-transposed bf16 Wt[N][KP] in global (L2-resident).
// LDS holds only the bf16 A strip. Branch-free templated epilogue.
// Block = 256 (8 waves): N==128 -> BM=16 (1x8), N==64 -> BM=32 (2x4).
// ---------------------------------------------------------------------------
template<bool BIAS, bool ACT, bool PRE, bool POST, bool ADD>
__global__ __launch_bounds__(256) void gemm_bf16(
    const float* __restrict__ A, const __bf16* __restrict__ Wt,
    const float* __restrict__ bias, const float* __restrict__ premul,
    const float* __restrict__ postmul, const float* __restrict__ addend,
    float* __restrict__ out, int M, int N, int K)
{
    extern __shared__ __bf16 Al[];       // [BM][KP]
    const int KP = (K + 31) & ~31;
    const int ntiles = N >> 4;           // 4 or 8
    const int BM = (8 / ntiles) << 4;    // 32 or 16
    const int tid = threadIdx.x;
    const int m0 = blockIdx.x * BM;

    for (int i = tid; i < BM * KP; i += 256) {
        int r = i / KP, c = i - r * KP;
        float v = 0.f;
        if (c < K) {
            size_t gi = (size_t)(m0 + r) * K + c;
            v = A[gi];
            if (PRE) v *= premul[gi];
        }
        Al[i] = f2bf(v);
    }
    __syncthreads();

    const int wave = tid >> 5, lane = tid & 31;
    const int row = lane & 15, hi = lane >> 4;
    const int mw = wave / ntiles, nw = wave - mw * ntiles;
    const int mb = mw << 4, n0 = nw << 4;
    const __bf16* Arow = Al + (mb + row) * KP + (hi ? 16 : 0);
    const __bf16* Brow = Wt + (size_t)(n0 + row) * KP + (hi ? 16 : 0);

    v8f acc = {};
    for (int kc = 0; kc < KP; kc += 32) {
        v8bf a0 = *(const v8bf*)(Arow + kc);        // K = kc + hi*16 + 0..7
        v8bf a1 = *(const v8bf*)(Arow + kc + 8);    // K = kc + hi*16 + 8..15
        v16bf a = __builtin_shufflevector(a0, a1, 0,1,2,3,4,5,6,7,8,9,10,11,12,13,14,15);
        v16bf b = *(const v16bf*)(Brow + kc);       // K = kc + hi*16 + 0..15
        acc = __builtin_amdgcn_wmma_f32_16x16x32_bf16(false, a, false, b,
                                                      (short)0, acc, false, false);
    }

    const int gn = n0 + row;
    float bv = BIAS ? bias[gn] : 0.f;
#pragma unroll
    for (int r = 0; r < 8; ++r) {
        int gm = m0 + mb + r + (hi << 3);
        size_t o = (size_t)gm * N + gn;
        float v = acc[r];
        if (BIAS) v += bv;
        if (ACT)  v = silu_f(v);
        if (POST) v *= postmul[o];
        if (ADD)  v += addend[o];
        out[o] = v;
    }
}

// ---------------------------------------------------------------------------
// Bilinear + fused segment-sum scatter:
//   tt[w,i] = sum_{j,l} t[w,j]*x[w,l]*Wbil[i,j,l];  out[idx_ji[w], i] += tt[w,i]
// All 4 N-tiles per wave: A-fragment built once feeds 4 WMMAs; TQ/XQT read 1x.
// grid = T/128, block = 256 (8 waves, one 16-row w-tile each).
// ---------------------------------------------------------------------------
__global__ __launch_bounds__(256) void bilinear_scatter(
    const float* __restrict__ TQ,   // [T,64]
    const float* __restrict__ XQT,  // [T,64]
    const __bf16* __restrict__ Wb,  // [64,64,64] bf16 (i,j,l)
    const int* __restrict__ idx_ji,
    float* __restrict__ out,        // [E,64] accum
    int Tn)
{
    const int wave = threadIdx.x >> 5, lane = threadIdx.x & 31;
    const int row = lane & 15, hi = lane >> 4;
    const int w0 = (blockIdx.x * 8 + wave) << 4;
    if (w0 >= Tn) return;

    float treg[64];
#pragma unroll
    for (int j = 0; j < 64; ++j) treg[j] = TQ[(size_t)(w0 + row) * 64 + j];
    float xreg[32];
#pragma unroll
    for (int lh = 0; lh < 2; ++lh)
#pragma unroll
        for (int h = 0; h < 16; ++h) {
            int kx = lh * 32 + (hi ? 16 : 0) + h;
            xreg[lh * 16 + h] = XQT[(size_t)(w0 + row) * 64 + kx];
        }

    const __bf16* Wrow = Wb + ((size_t)row << 12) + (hi << 4);  // + i_local*4096 + hi*16
    v8f acc0 = {}, acc1 = {}, acc2 = {}, acc3 = {};
    for (int j = 0; j < 64; ++j) {
        float tv = treg[j];
#pragma unroll
        for (int lh = 0; lh < 2; ++lh) {
            v16bf a;
#pragma unroll
            for (int h = 0; h < 16; ++h) a[h] = f2bf(tv * xreg[lh * 16 + h]);
            const __bf16* p = Wrow + (j << 6) + (lh << 5);
            acc0 = __builtin_amdgcn_wmma_f32_16x16x32_bf16(false, a, false,
                     *(const v16bf*)(p),                 (short)0, acc0, false, false);
            acc1 = __builtin_amdgcn_wmma_f32_16x16x32_bf16(false, a, false,
                     *(const v16bf*)(p + (16 << 12)),    (short)0, acc1, false, false);
            acc2 = __builtin_amdgcn_wmma_f32_16x16x32_bf16(false, a, false,
                     *(const v16bf*)(p + (32 << 12)),    (short)0, acc2, false, false);
            acc3 = __builtin_amdgcn_wmma_f32_16x16x32_bf16(false, a, false,
                     *(const v16bf*)(p + (48 << 12)),    (short)0, acc3, false, false);
        }
    }
#pragma unroll
    for (int r = 0; r < 8; ++r) {
        int w = w0 + r + (hi << 3);
        float* d = out + (size_t)idx_ji[w] * 64 + row;
        atomicAdd(d +  0, acc0[r]);
        atomicAdd(d + 16, acc1[r]);
        atomicAdd(d + 32, acc2[r]);
        atomicAdd(d + 48, acc3[r]);
    }
}

// ---------------------------------------------------------------------------
// Small per-edge rbf gate paths: gG=(rbf0_g@Wg1)@Wg2, gQ=(rbf0@Wq1)@Wq2, rbfH=rbf0@Wr
// ---------------------------------------------------------------------------
__global__ void rbf_paths(const float* __restrict__ rbf0, const float* __restrict__ rbf0g,
                          const float* __restrict__ Wg1, const float* __restrict__ Wg2,
                          const float* __restrict__ Wq1, const float* __restrict__ Wq2,
                          const float* __restrict__ Wr,
                          float* __restrict__ gG, float* __restrict__ gQ,
                          float* __restrict__ rbfH)
{
    int e = blockIdx.x, h = threadIdx.x;           // 128 threads
    __shared__ float rg[6], rq[6], mg[8], mq[8];
    if (h < 6) { rg[h] = rbf0g[e * 6 + h]; rq[h] = rbf0[e * 6 + h]; }
    __syncthreads();
    if (h < 8) {
        float a = 0.f, b = 0.f;
#pragma unroll
        for (int k = 0; k < 6; ++k) { a += rg[k] * Wg1[k * 8 + h]; b += rq[k] * Wq1[k * 8 + h]; }
        mg[h] = a; mq[h] = b;
    }
    __syncthreads();
    float a = 0.f, b = 0.f, c = 0.f;
#pragma unroll
    for (int j = 0; j < 8; ++j) { a += mg[j] * Wg2[j * 128 + h]; b += mq[j] * Wq2[j * 128 + h]; }
#pragma unroll
    for (int k = 0; k < 6; ++k) c += rq[k] * Wr[k * 128 + h];
    size_t o = (size_t)e * 128 + h;
    gG[o] = a; gQ[o] = b; rbfH[o] = c;
}

// ---------------------------------------------------------------------------
// segment_sum(src[idx_kj], idx_ji):  dst[idx_ji[w],:] += src[idx_kj[w],:]  (I=64)
// ---------------------------------------------------------------------------
__global__ void gather_scatter(const float* __restrict__ src, const int* __restrict__ ikj,
                               const int* __restrict__ iji, float* __restrict__ dst, int Tn)
{
    int g = blockIdx.x * blockDim.x + threadIdx.x;
    int w = g >> 4, c = (g & 15) << 2;
    if (w >= Tn) return;
    const float4 v = *(const float4*)(src + (size_t)ikj[w] * 64 + c);
    float* d = dst + (size_t)iji[w] * 64 + c;
    atomicAdd(d + 0, v.x); atomicAdd(d + 1, v.y);
    atomicAdd(d + 2, v.z); atomicAdd(d + 3, v.w);
}

// ---------------------------------------------------------------------------
// xqT[w,:] = xqI[idx_kj[w],:] * ((sbf[w] @ Wq_sbf1) @ Wq_sbf2)   (4 triplets/block)
// ---------------------------------------------------------------------------
__global__ void sbf_gate(const float* __restrict__ sbf, const float* __restrict__ Ws1,
                         const float* __restrict__ Ws2, const float* __restrict__ xqI,
                         const int* __restrict__ ikj, float* __restrict__ out, int Tn)
{
    __shared__ float s[4][42];
    __shared__ float mid[4][8];
    int q = threadIdx.x >> 6, t = threadIdx.x & 63;
    int w = blockIdx.x * 4 + q;
    if (w < Tn && t < 42) s[q][t] = sbf[(size_t)w * 42 + t];
    __syncthreads();
    if (w < Tn && t < 8) {
        float m = 0.f;
        for (int k = 0; k < 42; ++k) m += s[q][k] * Ws1[k * 8 + t];
        mid[q][t] = m;
    }
    __syncthreads();
    if (w < Tn) {
        float o = 0.f;
#pragma unroll
        for (int j = 0; j < 8; ++j) o += mid[q][j] * Ws2[j * 64 + t];
        out[(size_t)w * 64 + t] = o * xqI[(size_t)ikj[w] * 64 + t];
    }
}

__global__ void conv_w_bf16(const float* __restrict__ W, __bf16* __restrict__ Wb, int n)
{
    int i = blockIdx.x * blockDim.x + threadIdx.x;
    if (i < n) Wb[i] = f2bf(W[i]);
}

__global__ void ew_add(const float4* __restrict__ a, const float4* __restrict__ b,
                       float4* __restrict__ o, int n4)
{
    int i = blockIdx.x * blockDim.x + threadIdx.x;
    if (i < n4) {
        float4 x = a[i], y = b[i];
        o[i] = make_float4(x.x + y.x, x.y + y.y, x.z + y.z, x.w + y.w);
    }
}

__global__ void ew_mul(const float4* __restrict__ a, const float4* __restrict__ b,
                       float4* __restrict__ o, int n4)
{
    int i = blockIdx.x * blockDim.x + threadIdx.x;
    if (i < n4) {
        float4 x = a[i], y = b[i];
        o[i] = make_float4(x.x * y.x, x.y * y.y, x.z * y.z, x.w * y.w);
    }
}

// ---------------------------------------------------------------------------
static void gemm(const float* A, const __bf16* Wt, const float* bias,
                 const float* premul, const float* postmul, const float* addend,
                 float* out, int M, int N, int K, int act, hipStream_t s)
{
    int ntiles = N >> 4;
    int BM = (8 / ntiles) << 4;
    int KP = (K + 31) & ~31;
    size_t lds = (size_t)BM * KP * 2;
    dim3 g(M / BM), b(256);
#define LNCH(BI, AC, PR, PO, AD) \
    gemm_bf16<BI, AC, PR, PO, AD><<<g, b, lds, s>>>(A, Wt, bias, premul, postmul, addend, out, M, N, K)
    if (act) {
        if (premul)        LNCH(false, true, true,  false, false);
        else if (postmul)  LNCH(true,  true, false, true,  false);
        else if (addend)   LNCH(true,  true, false, false, true);
        else if (bias)     LNCH(true,  true, false, false, false);
        else               LNCH(false, true, false, false, false);
    } else {
        LNCH(false, false, false, false, false);
    }
#undef LNCH
}

static void transpose_w(const float* W, __bf16* Wt, int K, int N, hipStream_t s)
{
    int KP = (K + 31) & ~31;
    int n = N * KP;
    conv_wt<<<(n + 255) / 256, 256, 0, s>>>(W, Wt, K, N, KP);
}

extern "C" void kernel_launch(void* const* d_in, const int* in_sizes, int n_in,
                              void* d_out, int out_size, void* d_ws, size_t ws_size,
                              hipStream_t stream)
{
    const float* x1      = (const float*)d_in[0];
    const float* rbf0    = (const float*)d_in[1];
    const float* sbf     = (const float*)d_in[2];
    const float* tfeat   = (const float*)d_in[3];
    const float* rbf0g   = (const float*)d_in[4];
    const float* Wg_rbf1 = (const float*)d_in[5];
    const float* Wg_rbf2 = (const float*)d_in[6];
    const float* Wg_down = (const float*)d_in[7];
    const float* Wg_up   = (const float*)d_in[8];
    const float* Wg_kj   = (const float*)d_in[9];
    const float* bg_kj   = (const float*)d_in[10];
    const float* Wg_ji   = (const float*)d_in[11];
    const float* bg_ji   = (const float*)d_in[12];
    const float* Wq_rbf1 = (const float*)d_in[13];
    const float* Wq_rbf2 = (const float*)d_in[14];
    const float* Wq_sbf1 = (const float*)d_in[15];
    const float* Wq_sbf2 = (const float*)d_in[16];
    const float* Wq_t1   = (const float*)d_in[17];
    const float* Wq_down = (const float*)d_in[18];
    const float* Wq_up   = (const float*)d_in[19];
    const float* Wbil    = (const float*)d_in[20];
    const float* W_skip  = (const float*)d_in[21];
    const float* b_skip  = (const float*)d_in[22];
    const float* W_rbf   = (const float*)d_in[23];
    const float* Wb1     = (const float*)d_in[24];
    const float* Bb1     = (const float*)d_in[25];
    const float* Wb2     = (const float*)d_in[26];
    const float* Bb2     = (const float*)d_in[27];
    const float* Wa1     = (const float*)d_in[28];
    const float* Ba1     = (const float*)d_in[29];
    const float* Wa2     = (const float*)d_in[30];
    const float* Ba2     = (const float*)d_in[31];
    const int* idx_kj    = (const int*)d_in[32];
    const int* idx_ji    = (const int*)d_in[33];

    const size_t EH = (size_t)E_ * H_;
    const size_t EI = (size_t)E_ * I_;
    const size_t TI = (size_t)T_ * I_;

    float* f    = (float*)d_ws;
    float* gG   = f; f += EH;            // reused as T3 once dead
    float* gQ   = f; f += EH;
    float* rbfH = f; f += EH;
    float* XJI  = f; f += EH;            // x_ji_g, later E1
    float* XKJ  = f; f += EH;            // x_kj_g
    float* T1   = f; f += EH;
    float* T2   = f; f += EH;
    float* XKI  = f; f += EI;            // xk (I=64), reused as xqI
    float* XKS  = f; f += EI;            // segment accumulators (reused)
    float* TQb  = f; f += TI;            // t_feat @ Wq_t1
    float* XQT  = f; f += TI;            // sbf-gated xq[idx_kj]
    __bf16* wb  = (__bf16*)f;
    __bf16* Wb16 = wb;  wb += 64 * 64 * 64;       // bilinear weight, bf16
    // pre-transposed bf16 weights [N][KP]
    __bf16* WTgji  = wb; wb += 128 * 128;
    __bf16* WTgkj  = wb; wb += 128 * 128;
    __bf16* WTskip = wb; wb += 128 * 128;
    __bf16* WTb1   = wb; wb += 128 * 128;
    __bf16* WTb2   = wb; wb += 128 * 128;
    __bf16* WTa1_0 = wb; wb += 128 * 128;
    __bf16* WTa1_1 = wb; wb += 128 * 128;
    __bf16* WTa2_0 = wb; wb += 128 * 128;
    __bf16* WTa2_1 = wb; wb += 128 * 128;
    __bf16* WTgdn  = wb; wb += 64 * 128;
    __bf16* WTqdn  = wb; wb += 64 * 128;
    __bf16* WTgup  = wb; wb += 128 * 64;
    __bf16* WTqup  = wb; wb += 128 * 64;
    __bf16* WTqt1  = wb; wb += 64 * 320;
    float* T3  = gG;
    float* XQI = XKI;
    float* XQS = XKS;
    float* e1  = (float*)d_out;
    float* e2  = e1 + EH;

    const int n4 = (int)(EH / 4);

    // ---- weight preprocessing (tiny) ----
    transpose_w(Wg_ji,  WTgji,  128, 128, stream);
    transpose_w(Wg_kj,  WTgkj,  128, 128, stream);
    transpose_w(W_skip, WTskip, 128, 128, stream);
    transpose_w(Wb1,    WTb1,   128, 128, stream);
    transpose_w(Wb2,    WTb2,   128, 128, stream);
    transpose_w(Wa1,               WTa1_0, 128, 128, stream);
    transpose_w(Wa1 + 128 * 128,   WTa1_1, 128, 128, stream);
    transpose_w(Wa2,               WTa2_0, 128, 128, stream);
    transpose_w(Wa2 + 128 * 128,   WTa2_1, 128, 128, stream);
    transpose_w(Wg_down, WTgdn, 128, 64, stream);
    transpose_w(Wq_down, WTqdn, 128, 64, stream);
    transpose_w(Wg_up,   WTgup, 64, 128, stream);
    transpose_w(Wq_up,   WTqup, 64, 128, stream);
    transpose_w(Wq_t1,   WTqt1, 294, 64, stream);
    conv_w_bf16<<<(64 * 64 * 64) / 256, 256, 0, stream>>>(Wbil, Wb16, 64 * 64 * 64);

    // ---- gates ----
    rbf_paths<<<E_, 128, 0, stream>>>(rbf0, rbf0g, Wg_rbf1, Wg_rbf2,
                                      Wq_rbf1, Wq_rbf2, W_rbf, gG, gQ, rbfH);
    // ---- update_eG ----
    gemm(x1, WTgji, bg_ji, nullptr, nullptr, nullptr, XJI, E_, 128, 128, 1, stream);
    gemm(x1, WTgkj, bg_kj, nullptr, gG, nullptr, T1, E_, 128, 128, 1, stream);   // gG dead after
    gemm(T1, WTgdn, nullptr, nullptr, nullptr, nullptr, XKI, E_, 64, 128, 1, stream);
    hipMemsetAsync(XKS, 0, EI * sizeof(float), stream);
    gather_scatter<<<(int)(((size_t)T_ * 16) / 256), 256, 0, stream>>>(XKI, idx_kj, idx_ji, XKS, T_);
    gemm(XKS, WTgup, nullptr, nullptr, nullptr, nullptr, XKJ, E_, 128, 64, 1, stream);
    ew_add<<<n4 / 256, 256, 0, stream>>>((const float4*)XJI, (const float4*)XKJ, (float4*)T1, n4);
    // res before (1 layer)
    gemm(T1, WTb1, Bb1, nullptr, nullptr, nullptr, T2, E_, 128, 128, 1, stream);
    gemm(T2, WTb2, Bb2, nullptr, nullptr, T1, T3, E_, 128, 128, 1, stream);
    // skip + x1
    gemm(T3, WTskip, b_skip, nullptr, nullptr, x1, T1, E_, 128, 128, 1, stream);
    // res after (2 layers), ping-pong T1 <-> T3
    gemm(T1, WTa1_0, Ba1,       nullptr, nullptr, nullptr, T2, E_, 128, 128, 1, stream);
    gemm(T2, WTa2_0, Ba2,       nullptr, nullptr, T1, T3, E_, 128, 128, 1, stream);
    gemm(T3, WTa1_1, Ba1 + 128, nullptr, nullptr, nullptr, T2, E_, 128, 128, 1, stream);
    gemm(T2, WTa2_1, Ba2 + 128, nullptr, nullptr, T3, T1, E_, 128, 128, 1, stream); // qmpg in T1
    // ---- update_eQ ----
    gemm(XKJ, WTqdn, nullptr, gQ, nullptr, nullptr, XQI, E_, 64, 128, 1, stream);
    sbf_gate<<<T_ / 4, 256, 0, stream>>>(sbf, Wq_sbf1, Wq_sbf2, XQI, idx_kj, XQT, T_);
    gemm(tfeat, WTqt1, nullptr, nullptr, nullptr, nullptr, TQb, T_, 64, 294, 0, stream);
    hipMemsetAsync(XQS, 0, EI * sizeof(float), stream);
    bilinear_scatter<<<T_ / 128, 256, 0, stream>>>(TQb, XQT, Wb16, idx_ji, XQS, T_);
    gemm(XQS, WTqup, nullptr, nullptr, nullptr, nullptr, T2, E_, 128, 64, 1, stream);
    ew_add<<<n4 / 256, 256, 0, stream>>>((const float4*)XJI, (const float4*)T2, (float4*)T3, n4);
    // res before (1 layer)
    gemm(T3, WTb1, Bb1, nullptr, nullptr, nullptr, T2, E_, 128, 128, 1, stream);
    gemm(T2, WTb2, Bb2, nullptr, nullptr, T3, XJI, E_, 128, 128, 1, stream);      // E1 -> XJI
    // skip + qmpg(T1)
    gemm(XJI, WTskip, b_skip, nullptr, nullptr, T1, T3, E_, 128, 128, 1, stream);
    // res after (2 layers): T3 -> XJI -> d_out
    gemm(T3, WTa1_0, Ba1,       nullptr, nullptr, nullptr, T2, E_, 128, 128, 1, stream);
    gemm(T2, WTa2_0, Ba2,       nullptr, nullptr, T3, XJI, E_, 128, 128, 1, stream);
    gemm(XJI, WTa1_1, Ba1 + 128, nullptr, nullptr, nullptr, T2, E_, 128, 128, 1, stream);
    gemm(T2, WTa2_1, Ba2 + 128, nullptr, nullptr, XJI, e1, E_, 128, 128, 1, stream);
    // e2 = (rbf0 @ W_rbf) * e1
    ew_mul<<<n4 / 256, 256, 0, stream>>>((const float4*)rbfH, (const float4*)e1, (float4*)e2, n4);

    (void)in_sizes; (void)n_in; (void)out_size; (void)ws_size;
}
